// BiLSTM_37864431681632
// MI455X (gfx1250) — matrix operand order repaired and checked by reading, not compiled, not run on hardware
//
#include <hip/hip_runtime.h>

// ---------------------------------------------------------------------------
// BiLSTM (B=1024, T=1024, H=64) for gfx1250 (MI455X).
//
// z.T = Whh @ h.T computed with v_wmma_f32_16x16x32_f16:
//   A (16x32 f16)  = Whh tile      -> loop-invariant, held in VGPRs
//   B (32x16 f16)  = h.T tile      -> reloaded from LDS each step (ds_load_b128)
//   C/D (16x16 f32)= z tile        -> init = bias + x*Wih (VALU fma)
// One workgroup = 128 threads (4 wave32) = one 16-batch tile, one direction,
// full T loop on-chip. FC output projection fused per step.
// ---------------------------------------------------------------------------

typedef __attribute__((ext_vector_type(16))) _Float16 v16h;
typedef __attribute__((ext_vector_type(8)))  float    v8f;

#define HDIM    64
#define TLEN    1024
#define BATCH   1024
#define BTILE   16
#define ZSTRIDE 17   // padded row stride (floats) -> conflict-free zbuf

__device__ __forceinline__ float fast_sigmoid(float v) {
    float e = __builtin_amdgcn_exp2f(v * -1.4426950408889634f); // e^-v
    return __builtin_amdgcn_rcpf(1.0f + e);
}

__device__ __forceinline__ float fast_tanh(float v) {
#if __has_builtin(__builtin_amdgcn_tanhf)
    return __builtin_amdgcn_tanhf(v);
#else
    // tanh(v) = 2*sigmoid(2v) - 1
    float e = __builtin_amdgcn_exp2f(v * -2.8853900817779268f);
    return 2.0f * __builtin_amdgcn_rcpf(1.0f + e) - 1.0f;
#endif
}

__global__ __launch_bounds__(128) void bilstm_wmma_kernel(
    const float* __restrict__ x,
    const float* __restrict__ Wih_f, const float* __restrict__ Whh_f,
    const float* __restrict__ bih_f, const float* __restrict__ bhh_f,
    const float* __restrict__ Wih_b, const float* __restrict__ Whh_b,
    const float* __restrict__ bih_b, const float* __restrict__ bhh_b,
    const float* __restrict__ fc_w,  const float* __restrict__ fc_b,
    float* __restrict__ dst_fwd, float* __restrict__ dst_bwd,
    int dir_base, int acc_mode)
{
    __shared__ __align__(32) _Float16 hbuf[BTILE * HDIM];       // h state [b][k], f16
    __shared__ float                  zbuf[4 * HDIM * ZSTRIDE]; // z.T [m][b], padded

    const int tid    = threadIdx.x;
    const int lane   = tid & 31;
    const int wave   = tid >> 5;          // gate: 0=i 1=f 2=g 3=o
    const int laneLo = lane & 15;
    const int laneHi = lane >> 4;         // 0/1 -> which half of the wave
    const int bbase  = blockIdx.x * BTILE;
    const int dir    = (int)blockIdx.y + dir_base;

    const float* Wih = dir ? Wih_b : Wih_f;
    const float* Whh = dir ? Whh_b : Whh_f;
    const float* bih = dir ? bih_b : bih_f;
    const float* bhh = dir ? bhh_b : bhh_f;
    const float  fcb0 = fc_b[0];

    // ---- loop-invariant A fragments: Whh rows (f32 -> f16) ------------------
    // A fragment k order used here: lanes 0-15 hold K {0..15}; lanes 16-31 hold
    // K {16..31} (contiguous, offset by laneHi*16; matches B).
    v16h afrag[4][2];
#pragma unroll
    for (int j = 0; j < 4; ++j) {
        const int m = wave * 64 + j * 16 + laneLo;
        const float* wrow = Whh + m * HDIM;
#pragma unroll
        for (int kc = 0; kc < 2; ++kc) {
            const int k0 = kc * 32 + laneHi * 16;
            const int k1 = kc * 32 + 8 + laneHi * 16;
#pragma unroll
            for (int e = 0; e < 8; ++e) {
                afrag[j][kc][e]     = (_Float16)wrow[k0 + e];
                afrag[j][kc][8 + e] = (_Float16)wrow[k1 + e];
            }
        }
    }

    // ---- loop-invariant accumulator-init fragments (bias, Wih column) ------
    // C 16x16 f32 layout: reg r -> row m = r + laneHi*8, col = laneLo (batch).
    v8f biasv[4], wihv[4];
#pragma unroll
    for (int j = 0; j < 4; ++j) {
#pragma unroll
        for (int r = 0; r < 8; ++r) {
            const int m = wave * 64 + j * 16 + r + laneHi * 8;
            biasv[j][r] = bih[m] + bhh[m];
            wihv[j][r]  = Wih[m];          // Wih is (4H, 1)
        }
    }

    // ---- elementwise-phase constants: thread owns (b = tid>>3, 8 h-elems) --
    const int be  = tid >> 3;
    const int hb0 = (tid & 7) * 8;
    float fcv[8];
    float cstate[8];
#pragma unroll
    for (int e = 0; e < 8; ++e) {
        fcv[e]    = fc_w[dir * HDIM + hb0 + e];
        cstate[e] = 0.0f;
    }

    // h0 = 0
#pragma unroll
    for (int e = 0; e < 8; ++e) hbuf[tid * 8 + e] = (_Float16)0.0f;
    __syncthreads();

    // ======================= sequential time loop ===========================
    for (int t = 0; t < TLEN; ++t) {
        const int xidx = dir ? (TLEN - 1 - t) : t;   // input position
        const float xv = x[(size_t)(bbase + laneLo) * TLEN + xidx];

        // B fragments (h.T, 32x16 f16): lane holds col n=laneLo,
        // K = kc*32 + laneHi*16 + {0..15} -> 32 contiguous bytes of hbuf.
        v16h bfrag[2];
#pragma unroll
        for (int kc = 0; kc < 2; ++kc) {
            const _Float16* hp = &hbuf[laneLo * HDIM + kc * 32 + laneHi * 16];
            bfrag[kc] = *reinterpret_cast<const v16h*>(hp);
        }

        // z = bias + x*Wih + Whh @ h
        v8f acc[4];
#pragma unroll
        for (int j = 0; j < 4; ++j) {
#pragma unroll
            for (int r = 0; r < 8; ++r)
                acc[j][r] = __builtin_fmaf(xv, wihv[j][r], biasv[j][r]);
        }
#pragma unroll
        for (int j = 0; j < 4; ++j) {
            acc[j] = __builtin_amdgcn_wmma_f32_16x16x32_f16(
                false, afrag[j][0], false, bfrag[0], (short)0, acc[j], false, false);
            acc[j] = __builtin_amdgcn_wmma_f32_16x16x32_f16(
                false, afrag[j][1], false, bfrag[1], (short)0, acc[j], false, false);
        }

        // scatter z.T tiles to LDS
#pragma unroll
        for (int j = 0; j < 4; ++j) {
#pragma unroll
            for (int r = 0; r < 8; ++r) {
                const int m = wave * 64 + j * 16 + r + laneHi * 8;
                zbuf[m * ZSTRIDE + laneLo] = acc[j][r];
            }
        }
        __syncthreads();

        // ---- gates + cell/hidden update + fused FC partial ----
        float op = 0.0f;
#pragma unroll
        for (int e = 0; e < 8; ++e) {
            const int hh = hb0 + e;
            const float zi = zbuf[(hh)             * ZSTRIDE + be];
            const float zf = zbuf[(HDIM + hh)      * ZSTRIDE + be];
            const float zg = zbuf[(2 * HDIM + hh)  * ZSTRIDE + be];
            const float zo = zbuf[(3 * HDIM + hh)  * ZSTRIDE + be];
            float c = fast_sigmoid(zf) * cstate[e] + fast_sigmoid(zi) * fast_tanh(zg);
            cstate[e] = c;
            const float hn = fast_sigmoid(zo) * fast_tanh(c);
            hbuf[be * HDIM + hh] = (_Float16)hn;
            op = __builtin_fmaf(hn, fcv[e], op);
        }

        // reduce 8 threads per batch row (lanes grouped by 8 within the wave)
        op += __shfl_xor(op, 4, 32);
        op += __shfl_xor(op, 2, 32);
        op += __shfl_xor(op, 1, 32);
        if ((lane & 7) == 0) {
            float* dst = dir ? dst_bwd : dst_fwd;
            const size_t oidx = (size_t)(bbase + be) * TLEN + t;
            if (acc_mode && dir == 0) dst[oidx] = dst[oidx] + op + fcb0;
            else                      dst[oidx] = op;
        }
        __syncthreads();   // hbuf ready for next step; zbuf reads complete
    }
}

__global__ void combine_kernel(float* __restrict__ out,
                               const float* __restrict__ pb,
                               const float* __restrict__ fcb, int n)
{
    const int i = blockIdx.x * blockDim.x + threadIdx.x;
    if (i < n) out[i] = out[i] + pb[i] + fcb[0];
}

extern "C" void kernel_launch(void* const* d_in, const int* in_sizes, int n_in,
                              void* d_out, int out_size, void* d_ws, size_t ws_size,
                              hipStream_t stream)
{
    (void)in_sizes; (void)n_in;
    const float* x     = (const float*)d_in[0];
    const float* Wih_f = (const float*)d_in[1];
    const float* Whh_f = (const float*)d_in[2];
    const float* bih_f = (const float*)d_in[3];
    const float* bhh_f = (const float*)d_in[4];
    const float* Wih_b = (const float*)d_in[5];
    const float* Whh_b = (const float*)d_in[6];
    const float* bih_b = (const float*)d_in[7];
    const float* bhh_b = (const float*)d_in[8];
    const float* fc_w  = (const float*)d_in[9];
    const float* fc_b  = (const float*)d_in[10];
    float* out = (float*)d_out;
    float* ws  = (float*)d_ws;

    const dim3 blk(128);
    const size_t need = (size_t)BATCH * TLEN * sizeof(float);

    if (ws_size >= need) {
        // both directions concurrently; fwd partial -> d_out, bwd -> d_ws
        dim3 grid(BATCH / BTILE, 2);
        bilstm_wmma_kernel<<<grid, blk, 0, stream>>>(
            x, Wih_f, Whh_f, bih_f, bhh_f, Wih_b, Whh_b, bih_b, bhh_b,
            fc_w, fc_b, out, ws, /*dir_base=*/0, /*acc_mode=*/0);
        const int n = out_size > 0 ? out_size : BATCH * TLEN;
        combine_kernel<<<(n + 255) / 256, 256, 0, stream>>>(out, ws, fc_b, n);
    } else {
        // sequential fallback through d_out: backward partial, then forward+bias
        dim3 grid(BATCH / BTILE, 1);
        bilstm_wmma_kernel<<<grid, blk, 0, stream>>>(
            x, Wih_f, Whh_f, bih_f, bhh_f, Wih_b, Whh_b, bih_b, bhh_b,
            fc_w, fc_b, out, out, /*dir_base=*/1, /*acc_mode=*/0);
        bilstm_wmma_kernel<<<grid, blk, 0, stream>>>(
            x, Wih_f, Whh_f, bih_f, bhh_f, Wih_b, Whh_b, bih_b, bhh_b,
            fc_w, fc_b, out, out, /*dir_base=*/0, /*acc_mode=*/1);
    }
}
